// RBLNGptOssTopKRouter_77111842832396
// MI455X (gfx1250) — matrix operand order repaired and checked by reading, not compiled, not run on hardware
//
#include <hip/hip_runtime.h>
#include <hip/hip_bf16.h>

typedef __attribute__((ext_vector_type(16))) __bf16 v16bf;
typedef __attribute__((ext_vector_type(8)))  float  v8f;
typedef __attribute__((ext_vector_type(4)))  float  v4f;

#define HIDDEN   2880
#define NEXP     128
#define NTOK     16384              // 4 * 4096
#define KSTEP    32
#define NKSTEPS  (HIDDEN / KSTEP)   // 90
#define KSPLIT   2
#define KPER     (NKSTEPS / KSPLIT) // 45
#define TILES_PER_WG 4              // 64 tokens / WG
#define LSTRIDE  132                // padded reduction row stride (floats)

// ---- LDS staging geometry (bytes) ----
// row: 32 k * 2B = 64B data + 16B pad => stride 20 banks, conflict-free b128 reads
#define ROWB    80
#define PLANEB  (NEXP * ROWB)        // 10240  (one of hi/lo)
#define SECB    (2 * PLANEB)         // 20480  (hi+lo for one K-half)
#define BUFB    (2 * SECB)           // 40960  (both K-halves)
#define STAGEB  (2 * BUFB)           // 81920  (double buffer)
#define BIASOFF STAGEB
#define HISTOFF (STAGEB + 512)
#define SMEMB   (STAGEB + 1024)      // 82944 bytes dynamic LDS

// ---------------------------------------------------------------------------
// Prep: split f32 weight into bf16 hi/lo planes; zero the expert histogram.
// ---------------------------------------------------------------------------
__global__ void router_prep_kernel(const float* __restrict__ w,
                                   __bf16* __restrict__ w_hi,
                                   __bf16* __restrict__ w_lo,
                                   int* __restrict__ counts) {
    int i = blockIdx.x * 256 + threadIdx.x;
    if (i < NEXP * HIDDEN) {
        float f   = w[i];
        __bf16 hi = (__bf16)f;
        w_hi[i]   = hi;
        w_lo[i]   = (__bf16)(f - (float)hi);
    }
    if (blockIdx.x == 0 && threadIdx.x < NEXP) counts[threadIdx.x] = 0;
}

// async 16B global -> LDS copy (GLOBAL_LOAD_ASYNC_TO_LDS_B128, ASYNCcnt)
__device__ __forceinline__ void async_b128_to_lds(unsigned ldsAddr, const void* gaddr) {
    asm volatile("global_load_async_to_lds_b128 %0, %1, off"
                 :
                 : "v"(ldsAddr), "v"(gaddr)
                 : "memory");
}

// ---------------------------------------------------------------------------
// Main: bf16x3 WMMA GEMM (logits) + top-4 + softmax + scatter + histogram.
// Grid: 256 blocks x 256 threads (8 wave32). tile = wave>>1, ksec = wave&1.
// B fragments staged per-WG into LDS via async copies, double-buffered.
// ---------------------------------------------------------------------------
__global__ void __launch_bounds__(256)
router_topk_kernel(const float* __restrict__ h,
                   const __bf16* __restrict__ w_hi,
                   const __bf16* __restrict__ w_lo,
                   const float* __restrict__ bias,
                   float* __restrict__ outScores,
                   int* __restrict__ outIdx,
                   int* __restrict__ outCnt) {
    extern __shared__ __align__(16) char smem[];
    float* smL    = (float*)smem;              // aliases staging area after loop
    float* smbias = (float*)(smem + BIASOFF);
    int*   smhist = (int*)(smem + HISTOFF);

    const int lane   = threadIdx.x & 31;
    const int wave   = threadIdx.x >> 5;
    const int tile   = wave >> 1;
    const int ksec   = wave & 1;
    const int laneN  = lane & 15;
    const int hiHalf = lane >> 4;           // 0: lanes 0-15, 1: lanes 16-31
    const int koffA  = hiHalf * 16;         // A frag K-group offset (elements)
    const int koffB  = hiHalf * 16;         // B frag K-group offset (elements)
    const int tokenBase = blockIdx.x * (TILES_PER_WG * 16);
    const int row    = tokenBase + tile * 16 + laneN;

    if (threadIdx.x < NEXP) {
        smbias[threadIdx.x] = bias[threadIdx.x];
        smhist[threadIdx.x] = 0;
    }

    // ---- async stage of one 32-wide K chunk (both K-halves, hi+lo) ----
    // 32 KB per stage = 2048 x 16B chunks = 8 per thread.
    auto stage = [&](int ksIdx, int bufSel) {
#pragma unroll
        for (int j = 0; j < 8; ++j) {
            int li    = j * 256 + (int)threadIdx.x;   // 0..2047
            int sec   = li >> 10;                     // K-half
            int plane = (li >> 9) & 1;                // 0: hi, 1: lo
            int e     = (li >> 2) & 127;              // expert row
            int q     = li & 3;                       // 16B quarter of 64B row
            const __bf16* wp = plane ? w_lo : w_hi;
            const void* g = (const void*)((const char*)(wp + (size_t)e * HIDDEN
                              + sec * (KPER * KSTEP) + ksIdx * KSTEP) + q * 16);
            unsigned rel = (unsigned)(bufSel * BUFB + sec * SECB + plane * PLANEB
                                      + e * ROWB + q * 16);
            unsigned l = (unsigned)(size_t)(smem + rel);   // low 32 bits = LDS offset
            async_b128_to_lds(l, g);
        }
    };

    v8f acc[8] = {};   // 8 expert tiles x 16x16 f32 accumulators

    const float* hrow = h + (size_t)row * HIDDEN;
    const int kbase   = ksec * (KPER * KSTEP);   // 0 or 1440 (A addressing)

    stage(0, 0);
    for (int ks = 0; ks < KPER; ++ks) {
        __syncthreads();   // everyone done reading the buffer we overwrite next
        if (ks + 1 < KPER) {
            stage(ks + 1, (ks + 1) & 1);
            asm volatile("s_wait_asynccnt 8" ::: "memory");   // stage(ks) complete
        } else {
            asm volatile("s_wait_asynccnt 0" ::: "memory");
        }
        __syncthreads();   // stage(ks) visible to all waves

        const int kk = kbase + ks * KSTEP;
        __builtin_prefetch(hrow + kk + 8 * KSTEP, 0, 1);   // global_prefetch

        // ---- A fragment: 16 f32 -> bf16 hi/lo split (16-bit A layout) ----
        const v4f* p = (const v4f*)(hrow + kk + koffA);
        const v4f* q = (const v4f*)(hrow + kk + 8 + koffA);
        v4f f0 = p[0], f1 = p[1], f2 = q[0], f3 = q[1];
        union { v16bf v; __bf16 e[16]; } ahi, alo;
#pragma unroll
        for (int i = 0; i < 4; ++i) {
            float a = f0[i]; __bf16 a_h = (__bf16)a;
            ahi.e[i]      = a_h; alo.e[i]      = (__bf16)(a - (float)a_h);
            float b = f1[i]; __bf16 b_h = (__bf16)b;
            ahi.e[4 + i]  = b_h; alo.e[4 + i]  = (__bf16)(b - (float)b_h);
            float c = f2[i]; __bf16 c_h = (__bf16)c;
            ahi.e[8 + i]  = c_h; alo.e[8 + i]  = (__bf16)(c - (float)c_h);
            float d = f3[i]; __bf16 d_h = (__bf16)d;
            ahi.e[12 + i] = d_h; alo.e[12 + i] = (__bf16)(d - (float)d_h);
        }

        // ---- 8 expert tiles from LDS: bf16x3 accumulation ----
        const unsigned bBase = (unsigned)((ks & 1) * BUFB + ksec * SECB);
#pragma unroll
        for (int et = 0; et < 8; ++et) {
            unsigned ro = bBase + (unsigned)((et * 16 + laneN) * ROWB + koffB * 2);
            v16bf bhi = *(const v16bf*)(smem + ro);
            v16bf blo = *(const v16bf*)(smem + ro + PLANEB);
            acc[et] = __builtin_amdgcn_wmma_f32_16x16x32_bf16(
                false, alo.v, false, bhi, (short)0, acc[et], false, false);
            acc[et] = __builtin_amdgcn_wmma_f32_16x16x32_bf16(
                false, ahi.v, false, blo, (short)0, acc[et], false, false);
            acc[et] = __builtin_amdgcn_wmma_f32_16x16x32_bf16(
                false, ahi.v, false, bhi, (short)0, acc[et], false, false);
        }
    }

    // ---- K-split reduction through LDS (aliases staging buffers) ----
    __syncthreads();   // all staged-B reads done before aliasing
    if (ksec == 0) {
#pragma unroll
        for (int et = 0; et < 8; ++et) {
            union { v8f v; float f[8]; } ua; ua.v = acc[et];
#pragma unroll
            for (int r = 0; r < 8; ++r) {
                int m = r + (hiHalf << 3);              // C layout: M = r (+8)
                smL[(tile * 16 + m) * LSTRIDE + et * 16 + laneN] = ua.f[r];
            }
        }
    }
    __syncthreads();
    if (ksec == 1) {
#pragma unroll
        for (int et = 0; et < 8; ++et) {
            union { v8f v; float f[8]; } ua; ua.v = acc[et];
#pragma unroll
            for (int r = 0; r < 8; ++r) {
                int m = r + (hiHalf << 3);
                smL[(tile * 16 + m) * LSTRIDE + et * 16 + laneN] += ua.f[r];
            }
        }
    }
    __syncthreads();

    // ---- Top-4 + softmax + outputs: one thread per token ----
    if (threadIdx.x < TILES_PER_WG * 16) {
        const int t  = threadIdx.x;
        const int gt = tokenBase + t;
        const float* rowl = &smL[t * LSTRIDE];

        float v0 = -3.4e38f, v1 = -3.4e38f, v2 = -3.4e38f, v3 = -3.4e38f;
        int   i0 = 0, i1 = 0, i2 = 0, i3 = 0;
        for (int e = 0; e < NEXP; ++e) {
            float x = rowl[e] + smbias[e];
            if (x > v3) {
                if (x > v0)      { v3=v2;i3=i2; v2=v1;i2=i1; v1=v0;i1=i0; v0=x;i0=e; }
                else if (x > v1) { v3=v2;i3=i2; v2=v1;i2=i1; v1=x;i1=e; }
                else if (x > v2) { v3=v2;i3=i2; v2=x;i2=e; }
                else             { v3=x;i3=e; }
            }
        }
        float e1 = __expf(v1 - v0), e2 = __expf(v2 - v0), e3 = __expf(v3 - v0);
        float inv = 1.0f / (1.0f + e1 + e2 + e3);
        float s0 = inv, s1 = e1 * inv, s2 = e2 * inv, s3 = e3 * inv;

        float* so = outScores + (size_t)gt * NEXP;
        for (int e = 0; e < NEXP; ++e) {
            float val = 0.0f;
            if (e == i0) val = s0;
            if (e == i1) val = s1;
            if (e == i2) val = s2;
            if (e == i3) val = s3;
            so[e] = val;
        }
        int* oi = outIdx + (size_t)gt * 4;
        oi[0] = i0; oi[1] = i1; oi[2] = i2; oi[3] = i3;

        atomicAdd(&smhist[i0], 1);
        atomicAdd(&smhist[i1], 1);
        atomicAdd(&smhist[i2], 1);
        atomicAdd(&smhist[i3], 1);
    }
    __syncthreads();
    if (threadIdx.x < NEXP) {
        int c = smhist[threadIdx.x];
        if (c) atomicAdd(&outCnt[threadIdx.x], c);
    }
}

// ---------------------------------------------------------------------------
extern "C" void kernel_launch(void* const* d_in, const int* in_sizes, int n_in,
                              void* d_out, int out_size, void* d_ws, size_t ws_size,
                              hipStream_t stream) {
    const float* h    = (const float*)d_in[0];   // [4,4096,2880] f32
    const float* w    = (const float*)d_in[1];   // [128,2880]    f32
    const float* bias = (const float*)d_in[2];   // [128]         f32

    float* outScores = (float*)d_out;                              // [16384,128] f32
    int*   outIdx    = (int*)(outScores + (size_t)NTOK * NEXP);    // [16384,4]   i32
    int*   outCnt    = outIdx + (size_t)NTOK * 4;                  // [128]       i32

    __bf16* w_hi = (__bf16*)d_ws;                    // 128*2880 bf16
    __bf16* w_lo = w_hi + (size_t)NEXP * HIDDEN;     // 128*2880 bf16

    const int convElems = NEXP * HIDDEN;
    router_prep_kernel<<<(convElems + 255) / 256, 256, 0, stream>>>(w, w_hi, w_lo, outCnt);

    const int blocks = NTOK / (TILES_PER_WG * 16);   // 256
    router_topk_kernel<<<blocks, 256, SMEMB, stream>>>(h, w_hi, w_lo, bias,
                                                       outScores, outIdx, outCnt);
}
